// Generator_33036888441397
// MI455X (gfx1250) — matrix operand order repaired and checked by reading, not compiled, not run on hardware
//
#include <hip/hip_runtime.h>
#include <math.h>

// ---------------------------------------------------------------------------
// BRITS-style bidirectional 2-layer LSTM imputer for MI455X (gfx1250).
// B=512, T=64, D=16, H=1024.  Dominant work: 3x (512x4096x1024) GEMMs per
// step per direction -> v_wmma_f32_16x16x32_bf16 with f32 accumulate.
// 2x2 register tiling per wave: 4 WMMAs per 2A+2B fragment loads
// (16 FLOP per cache byte).  Weights bf16 (48MB -> resident in 192MB L2).
// Cell state c kept in f32; h stored as bf16 (WMMA A operand) + f32 (final).
// ---------------------------------------------------------------------------

typedef __bf16 bf16_t;
typedef __attribute__((ext_vector_type(16))) __bf16 v16bf;
typedef __attribute__((ext_vector_type(8)))  __bf16 v8bf;
typedef __attribute__((ext_vector_type(8)))  float  v8f;

#define BDIM 512
#define TDIM 64
#define DDIM 16
#define HDIM 1024
#define BH   (BDIM * HDIM)

__device__ __forceinline__ float sigf(float x) { return 1.0f / (1.0f + __expf(-x)); }

// ---- WMMA fragment loaders (contiguous k order, same in A and B) ----------
// A (16x32, MxK), row-major source with leading dim ld:
//   lane<16: M=lane,    slots 0-15 -> K=k0+0..15
//   lane>=16:M=lane-16, slots 0-15 -> K=k0+16..31
__device__ __forceinline__ v16bf load_a(const bf16_t* __restrict__ src,
                                        int row0, int k0, int ld) {
    int lane = threadIdx.x & 31;
    int half = lane >> 4;
    int m    = lane & 15;
    const bf16_t* p = src + (size_t)(row0 + m) * ld + k0 + half * 16;
    v8bf lo = *(const v8bf*)(p);        // K = k0 + 16*half + 0..7
    v8bf hi = *(const v8bf*)(p + 8);    // K = k0 + 16*half + 8..15
    v16bf a;
#pragma unroll
    for (int i = 0; i < 8; ++i) { a[i] = lo[i]; a[i + 8] = hi[i]; }
    return a;
}

// A with total K==16 (ld==16): upper half of K zero-padded.
__device__ __forceinline__ v16bf load_a_k16(const bf16_t* __restrict__ src,
                                            int row0) {
    int lane = threadIdx.x & 31;
    int half = lane >> 4;
    int m    = lane & 15;
    const bf16_t* p = src + (size_t)(row0 + m) * 16;
    v8bf lo = *(const v8bf*)(p);
    v8bf hi = *(const v8bf*)(p + 8);
    v16bf a;
#pragma unroll
    for (int i = 0; i < 8; ++i) { a[i] = half ? (bf16_t)0.0f : lo[i]; a[i + 8] = half ? (bf16_t)0.0f : hi[i]; }
    return a;
}

// B (32x16, KxN) where B[k][n] = W[wr0+n][k0+k], W row-major (rows x ld):
//   lane<16: N=lane,    slots -> K=k0+0..15 (contiguous in W's row)
//   lane>=16:N=lane-16, slots -> K=k0+16..31
__device__ __forceinline__ v16bf load_b(const bf16_t* __restrict__ W,
                                        int wr0, int k0, int ld) {
    int lane = threadIdx.x & 31;
    int half = lane >> 4;
    int n    = lane & 15;
    return *(const v16bf*)(W + (size_t)(wr0 + n) * ld + k0 + half * 16);
}

// B with total K==16 (ld==16): K>=16 half zero-padded.
__device__ __forceinline__ v16bf load_b_k16(const bf16_t* __restrict__ W,
                                            int wr0) {
    int lane = threadIdx.x & 31;
    int half = lane >> 4;
    int n    = lane & 15;
    v16bf b;
    if (half == 0) {
        b = *(const v16bf*)(W + (size_t)(wr0 + n) * 16);
    } else {
#pragma unroll
        for (int i = 0; i < 16; ++i) b[i] = (bf16_t)0.0f;
    }
    return b;
}

__device__ __forceinline__ v8f wmma_bf16(v16bf a, v16bf b, v8f c) {
    return __builtin_amdgcn_wmma_f32_16x16x32_bf16(false, a, false, b,
                                                   (short)0, c, false, false);
}

// ---- f32 -> bf16 weight conversion ----------------------------------------
__global__ void k_cvt(const float* __restrict__ src, bf16_t* __restrict__ dst, int n) {
    int i = blockIdx.x * blockDim.x + threadIdx.x;
    if (i < n) dst[i] = (bf16_t)src[i];
}

// ---- initial state: pre-loop layer-1 cell with zero inputs ----------------
__global__ void k_init(const float* __restrict__ bih1_f, const float* __restrict__ bhh1_f,
                       const float* __restrict__ bih1_b, const float* __restrict__ bhh1_b,
                       bf16_t* __restrict__ h0bf, float* __restrict__ c0,
                       bf16_t* __restrict__ h1bf, float* __restrict__ c1,
                       float* __restrict__ h1f) {
    int idx = blockIdx.x * blockDim.x + threadIdx.x;   // over B*H
    int dir = blockIdx.y;
    if (idx >= BH) return;
    int n = idx & (HDIM - 1);
    const float* bih = dir ? bih1_b : bih1_f;
    const float* bhh = dir ? bhh1_b : bhh1_f;
    float gi = bih[n]            + bhh[n];
    float gg = bih[2 * HDIM + n] + bhh[2 * HDIM + n];
    float go = bih[3 * HDIM + n] + bhh[3 * HDIM + n];
    float c1v = sigf(gi) * tanhf(gg);           // f-gate * c(=0) drops out
    float h1v = sigf(go) * tanhf(c1v);
    size_t si = (size_t)dir * BH + idx;
    h0bf[si] = (bf16_t)0.0f;
    c0[si]   = 0.0f;
    h1bf[si] = (bf16_t)h1v;
    c1[si]   = c1v;
    h1f[si]  = h1v;
}

// ---- cc = (1-m)*(h1 @ out_W^T + out_b) + m*x  (per direction) -------------
__global__ __launch_bounds__(32) void k_cc(
    const bf16_t* __restrict__ h1r,   // [dir][B*H]
    const bf16_t* __restrict__ woutB, // [16][1024] bf16
    const float* __restrict__ out_b,  // [16]
    const float* __restrict__ values, // [B][T][D]
    const float* __restrict__ masks,
    bf16_t* __restrict__ ccB,         // [dir][B*16]
    int step) {
    int dir  = blockIdx.y;
    int row0 = blockIdx.x * 16;
    const bf16_t* h1 = h1r + (size_t)dir * BH;
    v8f acc = {};
    for (int k0 = 0; k0 < HDIM; k0 += 32)
        acc = wmma_bf16(load_a(h1, row0, k0, HDIM), load_b(woutB, 0, k0, HDIM), acc);
    int t    = dir ? (TDIM - 1 - step) : step;
    int lane = threadIdx.x & 31;
    int half = lane >> 4, n = lane & 15;
    float ob = out_b[n];
    bf16_t* cc = ccB + (size_t)dir * (BDIM * DDIM);
#pragma unroll
    for (int r = 0; r < 8; ++r) {
        int brow = row0 + r + 8 * half;
        size_t xi = ((size_t)brow * TDIM + t) * DDIM + n;
        float m = masks[xi], x = values[xi];
        float v = (1.0f - m) * (acc[r] + ob) + m * x;
        cc[(size_t)brow * DDIM + n] = (bf16_t)v;
    }
}

// ---- layer-0 cell: gates = cc@Wih0^T + h0@Whh0^T + b; update h0,c0 --------
// Block = 32 rows x 32 hidden cols; 4 waves (one per gate quadrant),
// each wave computes a 2x2 grid of 16x16 WMMA tiles (4 WMMAs per 2A+2B loads).
__global__ __launch_bounds__(128) void k_cell0(
    const bf16_t* __restrict__ ccB,
    const bf16_t* __restrict__ h0r, bf16_t* __restrict__ h0w,
    float* __restrict__ c0,
    const bf16_t* __restrict__ wih0, const bf16_t* __restrict__ whh0,
    const float* __restrict__ bih_f, const float* __restrict__ bhh_f,
    const float* __restrict__ bih_b, const float* __restrict__ bhh_b) {
    const int dir  = blockIdx.z;
    const int row0 = blockIdx.x * 32;
    const int col0 = blockIdx.y * 32;
    const int w    = threadIdx.x >> 5;   // gate quadrant: 0=i 1=f 2=g 3=o
    const int lane = threadIdx.x & 31;
    const bf16_t* cc = ccB  + (size_t)dir * (BDIM * DDIM);
    const bf16_t* h0 = h0r  + (size_t)dir * BH;
    const bf16_t* Wi = wih0 + (size_t)dir * ((size_t)4 * HDIM * DDIM);
    const bf16_t* Wh = whh0 + (size_t)dir * ((size_t)4 * HDIM * HDIM);
    const float* bih = dir ? bih_b : bih_f;
    const float* bhh = dir ? bhh_b : bhh_f;
    const int wr0 = w * HDIM + col0;

    v8f acc00 = {}, acc01 = {}, acc10 = {}, acc11 = {};
    {   // K=16 input contribution (zero-padded to one K=32 WMMA)
        v16bf a0 = load_a_k16(cc, row0);
        v16bf a1 = load_a_k16(cc, row0 + 16);
        v16bf b0 = load_b_k16(Wi, wr0);
        v16bf b1 = load_b_k16(Wi, wr0 + 16);
        acc00 = wmma_bf16(a0, b0, acc00);
        acc01 = wmma_bf16(a0, b1, acc01);
        acc10 = wmma_bf16(a1, b0, acc10);
        acc11 = wmma_bf16(a1, b1, acc11);
    }
    for (int k0 = 0; k0 < HDIM; k0 += 32) {
        v16bf a0 = load_a(h0, row0,      k0, HDIM);
        v16bf a1 = load_a(h0, row0 + 16, k0, HDIM);
        v16bf b0 = load_b(Wh, wr0,      k0, HDIM);
        v16bf b1 = load_b(Wh, wr0 + 16, k0, HDIM);
        acc00 = wmma_bf16(a0, b0, acc00);
        acc01 = wmma_bf16(a0, b1, acc01);
        acc10 = wmma_bf16(a1, b0, acc10);
        acc11 = wmma_bf16(a1, b1, acc11);
    }

    __shared__ float gt[4][32][33];
    {
        int half = lane >> 4, n = lane & 15;
        float bias0 = bih[wr0 + n]      + bhh[wr0 + n];
        float bias1 = bih[wr0 + 16 + n] + bhh[wr0 + 16 + n];
#pragma unroll
        for (int i = 0; i < 8; ++i) {
            int rl = i + 8 * half;
            gt[w][rl][n]           = acc00[i] + bias0;
            gt[w][rl][16 + n]      = acc01[i] + bias1;
            gt[w][16 + rl][n]      = acc10[i] + bias0;
            gt[w][16 + rl][16 + n] = acc11[i] + bias1;
        }
    }
    __syncthreads();
#pragma unroll
    for (int e = 0; e < 8; ++e) {
        int idx = threadIdx.x + e * 128;     // 1024 elements / 128 threads
        int mr = idx >> 5, n = idx & 31;
        float gi = gt[0][mr][n], gf = gt[1][mr][n];
        float gg = gt[2][mr][n], go = gt[3][mr][n];
        size_t si = (size_t)dir * BH + (size_t)(row0 + mr) * HDIM + (col0 + n);
        float c  = c0[si];
        float cn = sigf(gf) * c + sigf(gi) * tanhf(gg);
        float hn = sigf(go) * tanhf(cn);
        c0[si]  = cn;
        h0w[si] = (bf16_t)hn;
    }
}

// ---- layer-1 cell: gates = h0@Wih1^T + h1@Whh1^T + b; update h1,c1 --------
__global__ __launch_bounds__(128) void k_cell1(
    const bf16_t* __restrict__ h0n,   // new h0 (written this step)
    const bf16_t* __restrict__ h1r, bf16_t* __restrict__ h1w,
    float* __restrict__ c1, float* __restrict__ h1f,
    const bf16_t* __restrict__ wih1, const bf16_t* __restrict__ whh1,
    const float* __restrict__ bih_f, const float* __restrict__ bhh_f,
    const float* __restrict__ bih_b, const float* __restrict__ bhh_b) {
    const int dir  = blockIdx.z;
    const int row0 = blockIdx.x * 32;
    const int col0 = blockIdx.y * 32;
    const int w    = threadIdx.x >> 5;
    const int lane = threadIdx.x & 31;
    const bf16_t* A1 = h0n  + (size_t)dir * BH;
    const bf16_t* A2 = h1r  + (size_t)dir * BH;
    const bf16_t* Wi = wih1 + (size_t)dir * ((size_t)4 * HDIM * HDIM);
    const bf16_t* Wh = whh1 + (size_t)dir * ((size_t)4 * HDIM * HDIM);
    const float* bih = dir ? bih_b : bih_f;
    const float* bhh = dir ? bhh_b : bhh_f;
    const int wr0 = w * HDIM + col0;

    v8f acc00 = {}, acc01 = {}, acc10 = {}, acc11 = {};
    for (int k0 = 0; k0 < HDIM; k0 += 32) {
        {
            v16bf a0 = load_a(A1, row0,      k0, HDIM);
            v16bf a1 = load_a(A1, row0 + 16, k0, HDIM);
            v16bf b0 = load_b(Wi, wr0,      k0, HDIM);
            v16bf b1 = load_b(Wi, wr0 + 16, k0, HDIM);
            acc00 = wmma_bf16(a0, b0, acc00);
            acc01 = wmma_bf16(a0, b1, acc01);
            acc10 = wmma_bf16(a1, b0, acc10);
            acc11 = wmma_bf16(a1, b1, acc11);
        }
        {
            v16bf a0 = load_a(A2, row0,      k0, HDIM);
            v16bf a1 = load_a(A2, row0 + 16, k0, HDIM);
            v16bf b0 = load_b(Wh, wr0,      k0, HDIM);
            v16bf b1 = load_b(Wh, wr0 + 16, k0, HDIM);
            acc00 = wmma_bf16(a0, b0, acc00);
            acc01 = wmma_bf16(a0, b1, acc01);
            acc10 = wmma_bf16(a1, b0, acc10);
            acc11 = wmma_bf16(a1, b1, acc11);
        }
    }

    __shared__ float gt[4][32][33];
    {
        int half = lane >> 4, n = lane & 15;
        float bias0 = bih[wr0 + n]      + bhh[wr0 + n];
        float bias1 = bih[wr0 + 16 + n] + bhh[wr0 + 16 + n];
#pragma unroll
        for (int i = 0; i < 8; ++i) {
            int rl = i + 8 * half;
            gt[w][rl][n]           = acc00[i] + bias0;
            gt[w][rl][16 + n]      = acc01[i] + bias1;
            gt[w][16 + rl][n]      = acc10[i] + bias0;
            gt[w][16 + rl][16 + n] = acc11[i] + bias1;
        }
    }
    __syncthreads();
#pragma unroll
    for (int e = 0; e < 8; ++e) {
        int idx = threadIdx.x + e * 128;
        int mr = idx >> 5, n = idx & 31;
        float gi = gt[0][mr][n], gf = gt[1][mr][n];
        float gg = gt[2][mr][n], go = gt[3][mr][n];
        size_t si = (size_t)dir * BH + (size_t)(row0 + mr) * HDIM + (col0 + n);
        float c  = c1[si];
        float cn = sigf(gf) * c + sigf(gi) * tanhf(gg);
        float hn = sigf(go) * tanhf(cn);
        c1[si]  = cn;
        h1w[si] = (bf16_t)hn;
        h1f[si] = hn;
    }
}

// ---- finalize: h = 0.5*(hf1 + flip_H(hb1)); outputs (h, imputations, imputed)
__global__ void k_final(const float* __restrict__ h1f,  // [dir][B*H]
                        const float* __restrict__ values,
                        const float* __restrict__ masks,
                        float* __restrict__ out) {
    int idx = blockIdx.x * blockDim.x + threadIdx.x;
    if (idx >= BH) return;
    int b = idx >> 10;          // / HDIM
    int j = idx & (HDIM - 1);
    float hv = 0.5f * (h1f[idx] + h1f[(size_t)BH + (size_t)b * HDIM + (HDIM - 1 - j)]);
    out[idx]          = hv;     // h
    out[BH + idx]     = hv;     // imputations (reshape of h: j = t*D + d)
    float m = masks[idx], v = values[idx];
    out[2 * BH + idx] = hv * (1.0f - m) + v * m;   // imputed
}

// ---------------------------------------------------------------------------
extern "C" void kernel_launch(void* const* d_in, const int* in_sizes, int n_in,
                              void* d_out, int out_size, void* d_ws, size_t ws_size,
                              hipStream_t stream) {
    (void)in_sizes; (void)n_in; (void)out_size; (void)ws_size;
    const float* values  = (const float*)d_in[0];
    const float* masks   = (const float*)d_in[1];
    const float* fw_Wih0 = (const float*)d_in[2];
    const float* fw_Whh0 = (const float*)d_in[3];
    const float* fw_bih0 = (const float*)d_in[4];
    const float* fw_bhh0 = (const float*)d_in[5];
    const float* fw_Wih1 = (const float*)d_in[6];
    const float* fw_Whh1 = (const float*)d_in[7];
    const float* fw_bih1 = (const float*)d_in[8];
    const float* fw_bhh1 = (const float*)d_in[9];
    const float* bw_Wih0 = (const float*)d_in[10];
    const float* bw_Whh0 = (const float*)d_in[11];
    const float* bw_bih0 = (const float*)d_in[12];
    const float* bw_bhh0 = (const float*)d_in[13];
    const float* bw_Wih1 = (const float*)d_in[14];
    const float* bw_Whh1 = (const float*)d_in[15];
    const float* bw_bih1 = (const float*)d_in[16];
    const float* bw_bhh1 = (const float*)d_in[17];
    const float* out_W   = (const float*)d_in[18];
    const float* out_b   = (const float*)d_in[19];
    // d_in[20..23] (fcb/fce) are dead code in the reference.

    // ---- workspace carve-out (256B aligned slots) ----
    char* ws = (char*)d_ws;
    size_t off = 0;
    auto alloc = [&](size_t bytes) -> char* {
        char* p = ws + off;
        off = (off + bytes + 255) & ~(size_t)255;
        return p;
    };
    const size_t W_BIG = (size_t)4 * HDIM * HDIM;   // 4096*1024 elements
    const size_t W_IH0 = (size_t)4 * HDIM * DDIM;   // 4096*16

    bf16_t* wOut = (bf16_t*)alloc((size_t)DDIM * HDIM * 2);
    bf16_t* wIh0 = (bf16_t*)alloc(2 * W_IH0 * 2);
    bf16_t* wHh0 = (bf16_t*)alloc(2 * W_BIG * 2);
    bf16_t* wIh1 = (bf16_t*)alloc(2 * W_BIG * 2);
    bf16_t* wHh1 = (bf16_t*)alloc(2 * W_BIG * 2);
    bf16_t* ccB  = (bf16_t*)alloc((size_t)2 * BDIM * DDIM * 2);
    bf16_t* h0bf[2], *h1bf[2];
    h0bf[0] = (bf16_t*)alloc((size_t)2 * BH * 2);
    h0bf[1] = (bf16_t*)alloc((size_t)2 * BH * 2);
    h1bf[0] = (bf16_t*)alloc((size_t)2 * BH * 2);
    h1bf[1] = (bf16_t*)alloc((size_t)2 * BH * 2);
    float* c0  = (float*)alloc((size_t)2 * BH * 4);
    float* c1  = (float*)alloc((size_t)2 * BH * 4);
    float* h1f = (float*)alloc((size_t)2 * BH * 4);

    // ---- convert weights to bf16 (L2-resident thereafter) ----
    auto cvt = [&](const float* s, bf16_t* d, size_t n) {
        k_cvt<<<dim3((unsigned)((n + 255) / 256)), 256, 0, stream>>>(s, d, (int)n);
    };
    cvt(out_W,   wOut,          (size_t)DDIM * HDIM);
    cvt(fw_Wih0, wIh0,          W_IH0);
    cvt(bw_Wih0, wIh0 + W_IH0,  W_IH0);
    cvt(fw_Whh0, wHh0,          W_BIG);
    cvt(bw_Whh0, wHh0 + W_BIG,  W_BIG);
    cvt(fw_Wih1, wIh1,          W_BIG);
    cvt(bw_Wih1, wIh1 + W_BIG,  W_BIG);
    cvt(fw_Whh1, wHh1,          W_BIG);
    cvt(bw_Whh1, wHh1 + W_BIG,  W_BIG);

    // ---- initial states ----
    k_init<<<dim3((BH + 255) / 256, 2), 256, 0, stream>>>(
        fw_bih1, fw_bhh1, bw_bih1, bw_bhh1,
        h0bf[0], c0, h1bf[0], c1, h1f);

    // ---- time scan (ping-pong h buffers to avoid cross-block RAW) ----
    for (int s = 0; s < TDIM; ++s) {
        int rb = s & 1, wb = rb ^ 1;
        k_cc<<<dim3(BDIM / 16, 2), 32, 0, stream>>>(
            h1bf[rb], wOut, out_b, values, masks, ccB, s);
        k_cell0<<<dim3(BDIM / 32, HDIM / 32, 2), 128, 0, stream>>>(
            ccB, h0bf[rb], h0bf[wb], c0, wIh0, wHh0,
            fw_bih0, fw_bhh0, bw_bih0, bw_bhh0);
        k_cell1<<<dim3(BDIM / 32, HDIM / 32, 2), 128, 0, stream>>>(
            h0bf[wb], h1bf[rb], h1bf[wb], c1, h1f, wIh1, wHh1,
            fw_bih1, fw_bhh1, bw_bih1, bw_bhh1);
    }

    // ---- outputs ----
    k_final<<<(BH + 255) / 256, 256, 0, stream>>>(h1f, values, masks, (float*)d_out);
}
